// EncoderDecoderAttention_4655744549079
// MI455X (gfx1250) — matrix operand order repaired and checked by reading, not compiled, not run on hardware
//
#include <hip/hip_runtime.h>

#define BB 8
#define NN 1024
#define DD 1024
#define EE 128
#define HH 16
#define HE 2048   // H*E

typedef unsigned short u16;
typedef __attribute__((ext_vector_type(16))) __bf16 bf16x16;
typedef __attribute__((ext_vector_type(8)))  float  f32x8;

#define NEG_BIG (-3.0e38f)

static __device__ __forceinline__ u16 f2bf(float f) {
  unsigned u = __float_as_uint(f);
  return (u16)((u + 0x7FFFu + ((u >> 16) & 1u)) >> 16);
}

// Build a 16-element bf16 fragment from two 16-byte chunks.
// A-operand (16x32, row r=lane%16, half=lane/16): chunks at k0+half*8 and k0+half*8+16.
// B-operand (32x16, col n=lane%16, half=lane/16): chunks at k0+half*8 and k0+half*8+16.
static __device__ __forceinline__ bf16x16 frag2(const u16* p0, const u16* p1) {
  union { uint4 q[2]; bf16x16 v; } u;
  u.q[0] = *(const uint4*)p0;
  u.q[1] = *(const uint4*)p1;
  return u.v;
}

static __device__ __forceinline__ f32x8 wmma_bf16(bf16x16 a, bf16x16 b, f32x8 c) {
  return __builtin_amdgcn_wmma_f32_16x16x32_bf16(false, a, false, b, (short)0, c,
                                                 false, false);
}

// ---------------- elementwise f32 -> bf16 ----------------
__global__ void k_cvt(const float* __restrict__ in, u16* __restrict__ out, size_t n) {
  size_t i = (size_t)blockIdx.x * blockDim.x + threadIdx.x;
  size_t st = (size_t)gridDim.x * blockDim.x;
  for (; i < n; i += st) out[i] = f2bf(in[i]);
}

// ---------------- [G][R][C] f32  ->  [G][C][R] bf16 (transpose) ----------------
__global__ void k_cvt_t(const float* __restrict__ in, u16* __restrict__ out,
                        int G, int R, int C) {
  size_t total = (size_t)G * R * C;
  size_t st = (size_t)gridDim.x * blockDim.x;
  size_t rc = (size_t)R * C;
  for (size_t i = (size_t)blockIdx.x * blockDim.x + threadIdx.x; i < total; i += st) {
    size_t g = i / rc;
    size_t rem = i - g * rc;
    size_t c = rem / (size_t)R;
    size_t r = rem - c * (size_t)R;
    out[i] = f2bf(in[g * rc + r * (size_t)C + c]);
  }
}

// ---------------- generic batched GEMM: C[M x Nn] = A[M x K] * Bt[Nn x K]^T ----
// One wave computes a 32x32 block (2x2 WMMA tiles), K-loop ping-pong pipelined.
// Requires M%32==0, Nn%32==0, K%64==0 (true for all uses here).
template <bool OBF>
__global__ void k_gemm(const u16* __restrict__ A, size_t sA, int aDiv, int aMod, int lda,
                       const u16* __restrict__ Bt, size_t sB, int bDiv, int bMod, int ldb,
                       void* __restrict__ Cp, size_t sC, int ldc,
                       int M, int Nn, int K) {
  int lane = threadIdx.x & 31;
  int w = blockIdx.x * (blockDim.x >> 5) + (threadIdx.x >> 5);
  int tilesN = Nn >> 5;
  int total = (M >> 5) * tilesN;
  if (w >= total) return;
  int tm = w / tilesN, tn = w - tm * tilesN;
  int batch = blockIdx.y;

  const u16* Ab = A + sA * (size_t)((batch / aDiv) % aMod);
  const u16* Bb = Bt + sB * (size_t)((batch / bDiv) % bMod);
  int r = lane & 15, hf = lane >> 4;
  const u16* pa0 = Ab + (size_t)(tm * 32 + r) * lda + hf * 8;
  const u16* pa1 = pa0 + (size_t)16 * lda;
  const u16* pb0 = Bb + (size_t)(tn * 32 + r) * ldb + hf * 8;
  const u16* pb1 = pb0 + (size_t)16 * ldb;

  f32x8 acc00 = {}, acc01 = {}, acc10 = {}, acc11 = {};

  // ping-pong register buffers (x: even K-step, y: odd K-step)
  bf16x16 xa0 = frag2(pa0, pa0 + 16), xa1 = frag2(pa1, pa1 + 16);
  bf16x16 xb0 = frag2(pb0, pb0 + 16),  xb1 = frag2(pb1, pb1 + 16);
  bf16x16 ya0, ya1, yb0, yb1;

  for (int k0 = 0; k0 < K; k0 += 64) {
    int k1 = k0 + 32;
    // issue loads for the odd step, then compute on the even step
    ya0 = frag2(pa0 + k1, pa0 + k1 + 16);
    ya1 = frag2(pa1 + k1, pa1 + k1 + 16);
    yb0 = frag2(pb0 + k1, pb0 + k1 + 16);
    yb1 = frag2(pb1 + k1, pb1 + k1 + 16);
    __builtin_prefetch(pa0 + k1 + 32, 0, 0);
    __builtin_prefetch(pb0 + k1 + 32, 0, 0);
    acc00 = wmma_bf16(xa0, xb0, acc00);
    acc01 = wmma_bf16(xa0, xb1, acc01);
    acc10 = wmma_bf16(xa1, xb0, acc10);
    acc11 = wmma_bf16(xa1, xb1, acc11);
    int k2 = k0 + 64;
    if (k2 < K) {  // issue loads for the next even step, compute on the odd step
      xa0 = frag2(pa0 + k2, pa0 + k2 + 16);
      xa1 = frag2(pa1 + k2, pa1 + k2 + 16);
      xb0 = frag2(pb0 + k2, pb0 + k2 + 16);
      xb1 = frag2(pb1 + k2, pb1 + k2 + 16);
    }
    acc00 = wmma_bf16(ya0, yb0, acc00);
    acc01 = wmma_bf16(ya0, yb1, acc01);
    acc10 = wmma_bf16(ya1, yb0, acc10);
    acc11 = wmma_bf16(ya1, yb1, acc11);
  }

  int row0 = tm * 32, col0 = tn * 32;
  if constexpr (OBF) {
    u16* C = (u16*)Cp + sC * (size_t)batch;
#pragma unroll
    for (int j = 0; j < 8; ++j) {
      size_t ro0 = (size_t)(row0 + j + hf * 8) * ldc;
      size_t ro1 = (size_t)(row0 + 16 + j + hf * 8) * ldc;
      C[ro0 + col0 + r]      = f2bf(acc00[j]);
      C[ro0 + col0 + 16 + r] = f2bf(acc01[j]);
      C[ro1 + col0 + r]      = f2bf(acc10[j]);
      C[ro1 + col0 + 16 + r] = f2bf(acc11[j]);
    }
  } else {
    float* C = (float*)Cp + sC * (size_t)batch;
#pragma unroll
    for (int j = 0; j < 8; ++j) {
      size_t ro0 = (size_t)(row0 + j + hf * 8) * ldc;
      size_t ro1 = (size_t)(row0 + 16 + j + hf * 8) * ldc;
      C[ro0 + col0 + r]      = acc00[j];
      C[ro0 + col0 + 16 + r] = acc01[j];
      C[ro1 + col0 + r]      = acc10[j];
      C[ro1 + col0 + 16 + r] = acc11[j];
    }
  }
}

// ---------------- pass A: column (query-axis) softmax stats ----------------
// For each (b,h,m): colmax[m] = max_{n<mc} s[n,m], colsum[m] = sum exp(s-colmax).
// Branch-free online update using a -3e38 sentinel (underflows to 0 in exp).
__global__ void k_colstats(const u16* __restrict__ Qb, const u16* __restrict__ Kb,
                           float* __restrict__ colmax, float* __restrict__ rcpsum,
                           const int* __restrict__ cur) {
  int lane = threadIdx.x & 31;
  int wg = blockIdx.x * 8 + (threadIdx.x >> 5);
  if (wg >= BB * HH * (NN / 16)) return;
  int bh = wg >> 6;      // NN/16 == 64
  int mt = wg & 63;
  int mc = *cur + 1; if (mc > NN - 1) mc = NN - 1;
  const float scale = 0.08838834764831843f;  // 1/sqrt(E)

  const u16* Q  = Qb + (size_t)bh * NN * EE;
  const u16* Kk = Kb + (size_t)bh * NN * EE;
  int r = lane & 15, hf = lane >> 4;

  // hoist the K-tile B fragments (loop invariant)
  bf16x16 kb[4];
  {
    const u16* pb = Kk + (size_t)(mt * 16 + r) * EE + hf * 8;
#pragma unroll
    for (int ks = 0; ks < 4; ++ks) kb[ks] = frag2(pb + ks * 32, pb + ks * 32 + 16);
  }

  float runM = NEG_BIG, runS = 0.0f;
  auto upd = [&](float v) {
    float nm = fmaxf(runM, v);
    runS = runS * __expf(runM - nm) + __expf(v - nm);
    runM = nm;
  };

  int ntiles = (mc + 15) >> 4;
  int nt = 0;
  for (; nt + 1 < ntiles; nt += 2) {  // two n-tiles per iteration
    const u16* paA = Q + (size_t)(nt * 16 + r) * EE + hf * 8;
    const u16* paB = paA + (size_t)16 * EE;
    bf16x16 qa[4], qb[4];
#pragma unroll
    for (int ks = 0; ks < 4; ++ks) {
      qa[ks] = frag2(paA + ks * 32, paA + ks * 32 + 16);
      qb[ks] = frag2(paB + ks * 32, paB + ks * 32 + 16);
    }
    f32x8 s0 = {}, s1 = {};
#pragma unroll
    for (int ks = 0; ks < 4; ++ks) {
      s0 = wmma_bf16(qa[ks], kb[ks], s0);
      s1 = wmma_bf16(qb[ks], kb[ks], s1);
    }
#pragma unroll
    for (int j = 0; j < 8; ++j) {
      int n0 = nt * 16 + j + hf * 8;
      upd((n0 < mc)      ? s0[j] * scale : NEG_BIG);
      upd((n0 + 16 < mc) ? s1[j] * scale : NEG_BIG);
    }
  }
  if (nt < ntiles) {  // odd remainder tile
    const u16* pa = Q + (size_t)(nt * 16 + r) * EE + hf * 8;
    f32x8 s0 = {};
#pragma unroll
    for (int ks = 0; ks < 4; ++ks)
      s0 = wmma_bf16(frag2(pa + ks * 32, pa + ks * 32 + 16), kb[ks], s0);
#pragma unroll
    for (int j = 0; j < 8; ++j) {
      int n0 = nt * 16 + j + hf * 8;
      upd((n0 < mc) ? s0[j] * scale : NEG_BIG);
    }
  }

  // merge the two lane-halves (same column m, different row groups)
  float oM = __shfl_xor(runM, 16, 32);
  float oS = __shfl_xor(runS, 16, 32);
  float nm = fmaxf(runM, oM);
  float S = runS * __expf(runM - nm) + oS * __expf(oM - nm);
  if (hf == 0) {
    int m = mt * 16 + r;
    colmax[(size_t)bh * NN + m] = nm;
    rcpsum[(size_t)bh * NN + m] = __frcp_rn(S);
  }
}

// ---------------- pass B: heads = P @ (V/colsum), P = exp(s-colmax) ----------
// Each wave: one (b,h,n-tile); accumulates full 16x128 heads rows; P goes
// C-layout -> LDS -> A-fragment layout. V B-fragments preloaded per m-chunk so
// their latency hides behind the score WMMAs and exp work.
__global__ void k_attn(const u16* __restrict__ Qb, const u16* __restrict__ Kb,
                       const u16* __restrict__ Vt,
                       const float* __restrict__ colmax, const float* __restrict__ rcpsum,
                       u16* __restrict__ multi, const int* __restrict__ cur) {
  __shared__ __align__(16) u16 pshm[8][16 * 32];
  int lane = threadIdx.x & 31;
  int wv = threadIdx.x >> 5;
  int wg = blockIdx.x * 8 + wv;
  if (wg >= BB * HH * 64) return;
  int bh = wg >> 6, nt = wg & 63;
  int b = bh >> 4, h = bh & 15;
  int mc = *cur + 1; if (mc > NN - 1) mc = NN - 1;
  int r = lane & 15, hf = lane >> 4;

  if (nt * 16 >= mc) {  // fully masked query rows -> zero output
#pragma unroll
    for (int j = 0; j < 8; ++j) {
      int n = nt * 16 + j + hf * 8;
      u16* dst = multi + ((size_t)b * NN + n) * HE + (size_t)h * EE + r;
#pragma unroll
      for (int et = 0; et < 8; ++et) dst[et * 16] = 0;
    }
    return;
  }

  const float scale = 0.08838834764831843f;
  const u16* Q  = Qb + (size_t)bh * NN * EE;
  const u16* Kk = Kb + (size_t)bh * NN * EE;
  const u16* V  = Vt + (size_t)bh * EE * NN;   // [E][N] (transposed V)
  const float* cm = colmax + (size_t)bh * NN;
  const float* rs = rcpsum + (size_t)bh * NN;

  // per-row additive mask bias (0 valid, -3e38 masked): exp -> 0, branch-free
  float rowbias[8];
#pragma unroll
  for (int j = 0; j < 8; ++j)
    rowbias[j] = (nt * 16 + j + hf * 8 < mc) ? 0.0f : NEG_BIG;

  f32x8 z = {};
  f32x8 acc[8];
#pragma unroll
  for (int et = 0; et < 8; ++et) acc[et] = z;

  const u16* paq = Q + (size_t)(nt * 16 + r) * EE + hf * 8;

  for (int m0 = 0; m0 < NN; m0 += 32) {
    // preload V' B-fragments for this m-chunk (overlaps with score WMMAs below)
    bf16x16 vf[8];
#pragma unroll
    for (int et = 0; et < 8; ++et) {
      const u16* pv = V + (size_t)(et * 16 + r) * NN + m0 + hf * 8;
      vf[et] = frag2(pv, pv + 16);
    }

    // two 16x16 score tiles -> P -> LDS (as a 16x32 bf16 A tile)
#pragma unroll
    for (int t = 0; t < 2; ++t) {
      const u16* pb = Kk + (size_t)(m0 + t * 16 + r) * EE + hf * 8;
      f32x8 s = z;
#pragma unroll
      for (int ks = 0; ks < 4; ++ks)
        s = wmma_bf16(frag2(paq + ks * 32, paq + ks * 32 + 16),
                      frag2(pb + ks * 32, pb + ks * 32 + 16), s);
      int m = m0 + t * 16 + r;
      float cmx = cm[m], rsm = rs[m];
#pragma unroll
      for (int j = 0; j < 8; ++j) {
        float p = __expf(s[j] * scale - cmx + rowbias[j]) * rsm;
        pshm[wv][(j + hf * 8) * 32 + t * 16 + r] = f2bf(p);
      }
    }
    asm volatile("s_wait_dscnt 0" ::: "memory");  // P stores visible to A-frag loads
    bf16x16 pfrag = frag2(&pshm[wv][r * 32 + hf * 8],
                          &pshm[wv][r * 32 + hf * 8 + 16]);
#pragma unroll
    for (int et = 0; et < 8; ++et) acc[et] = wmma_bf16(pfrag, vf[et], acc[et]);
  }

#pragma unroll
  for (int et = 0; et < 8; ++et)
#pragma unroll
    for (int j = 0; j < 8; ++j) {
      int n = nt * 16 + j + hf * 8;
      multi[((size_t)b * NN + n) * HE + (size_t)h * EE + et * 16 + r] = f2bf(acc[et][j]);
    }
}

extern "C" void kernel_launch(void* const* d_in, const int* in_sizes, int n_in,
                              void* d_out, int out_size, void* d_ws, size_t ws_size,
                              hipStream_t stream) {
  const float* x    = (const float*)d_in[0];
  const float* ctx  = (const float*)d_in[1];
  // d_in[2] attention_mask: unused by the reference module
  const float* wq   = (const float*)d_in[3];
  const float* wk   = (const float*)d_in[4];
  const float* wv   = (const float*)d_in[5];
  const float* wagg = (const float*)d_in[6];
  const int*   cur  = (const int*)d_in[7];
  float* out = (float*)d_out;

  char* ws = (char*)d_ws;
  size_t off = 0;
  auto carve = [&](size_t bytes) -> char* {
    char* p = ws + off;
    off += (bytes + 255) & ~(size_t)255;
    return p;
  };
  u16* xb   = (u16*)carve((size_t)BB * NN * DD * 2);
  u16* cb   = (u16*)carve((size_t)BB * NN * DD * 2);
  u16* wqT  = (u16*)carve((size_t)HH * EE * DD * 2);
  u16* wkT  = (u16*)carve((size_t)HH * EE * DD * 2);
  u16* wvT  = (u16*)carve((size_t)HH * EE * DD * 2);
  u16* waT  = (u16*)carve((size_t)DD * HE * 2);
  u16* Qb   = (u16*)carve((size_t)BB * HH * NN * EE * 2);
  u16* Kb   = (u16*)carve((size_t)BB * HH * NN * EE * 2);
  u16* Vt   = (u16*)carve((size_t)BB * HH * EE * NN * 2);
  u16* mlt  = (u16*)carve((size_t)BB * NN * HE * 2);
  float* cmx = (float*)carve((size_t)BB * HH * NN * 4);
  float* rsm = (float*)carve((size_t)BB * HH * NN * 4);

  // ---- precision conversion (+ weight transposes for B-operand layout) ----
  k_cvt<<<2048, 256, 0, stream>>>(x, xb, (size_t)BB * NN * DD);
  k_cvt<<<2048, 256, 0, stream>>>(ctx, cb, (size_t)BB * NN * DD);
  k_cvt_t<<<1024, 256, 0, stream>>>(wq, wqT, HH, DD, EE);
  k_cvt_t<<<1024, 256, 0, stream>>>(wk, wkT, HH, DD, EE);
  k_cvt_t<<<1024, 256, 0, stream>>>(wv, wvT, HH, DD, EE);
  k_cvt_t<<<1024, 256, 0, stream>>>(wagg, waT, 1, HE, DD);

  // ---- QKV projections (batched over bh = b*H+h); 32x32 block per wave ----
  dim3 g1(16, BB * HH);   // 128 waves / batch
  // Q[bh] (N x E) = x[b] (N x D) * wq[h]; Bt = wqT[h] (E x D)
  k_gemm<true><<<g1, 256, 0, stream>>>(xb, (size_t)NN * DD, HH, BB, DD,
                                       wqT, (size_t)EE * DD, 1, HH, DD,
                                       Qb, (size_t)NN * EE, EE, NN, EE, DD);
  k_gemm<true><<<g1, 256, 0, stream>>>(cb, (size_t)NN * DD, HH, BB, DD,
                                       wkT, (size_t)EE * DD, 1, HH, DD,
                                       Kb, (size_t)NN * EE, EE, NN, EE, DD);
  // Vt[bh] (E x N) = wvT[h] (E x D) * ctx[b]^T ; Bt = cb[b] (N x D)
  k_gemm<true><<<g1, 256, 0, stream>>>(wvT, (size_t)EE * DD, 1, HH, DD,
                                       cb, (size_t)NN * DD, HH, BB, DD,
                                       Vt, (size_t)EE * NN, NN, EE, NN, DD);

  // ---- attention: column-softmax stats, then fused P@V' into multi ----
  k_colstats<<<1024, 256, 0, stream>>>(Qb, Kb, cmx, rsm, cur);
  k_attn<<<1024, 256, 0, stream>>>(Qb, Kb, Vt, cmx, rsm, mlt, cur);

  // ---- output projection: out (B*N x D) = multi (B*N x HE) * w_agg ----
  dim3 gf(1024, 1);       // 8192 waves
  k_gemm<false><<<gf, 256, 0, stream>>>(mlt, 0, 1, 1, HE,
                                        waT, 0, 1, 1, HE,
                                        out, 0, DD, BB * NN, DD, HE);
}
